// ImportanceGetter_11828339933493
// MI455X (gfx1250) — matrix operand order repaired and checked by reading, not compiled, not run on hardware
//
#include <hip/hip_runtime.h>
#include <hip/hip_bf16.h>

typedef __attribute__((ext_vector_type(16))) __bf16 v16bf;
typedef __attribute__((ext_vector_type(8)))  __bf16 v8bf;
typedef __attribute__((ext_vector_type(8)))  float  v8f;

#define DDIM 768
#define D2   1536
#define PDIM 128
#define CDIM 512

__device__ __forceinline__ void split_bf16(float x, __bf16 &h, __bf16 &l) {
    h = (__bf16)x;
    l = (__bf16)(x - (float)h);
}

// A-operand fragment: rowp points at (row, k_tile_base). Lanes 0-15 hold K
// {0..15}, lanes 16-31 hold {16..31} (ksel = 0 or 16).
__device__ __forceinline__ v16bf load_a_frag(const __bf16* rowp, int ksel) {
    v8bf c0 = *(const v8bf*)(rowp + ksel);
    v8bf c1 = *(const v8bf*)(rowp + 8 + ksel);
    return __builtin_shufflevector(c0, c1, 0,1,2,3,4,5,6,7,8,9,10,11,12,13,14,15);
}

#define WMMA_BF16(A, B, C) \
    __builtin_amdgcn_wmma_f32_16x16x32_bf16(false, (A), false, (B), (short)0, (C), false, false)

// ---------------- prep: layernorm rows of [rows, 768] ----------------
__global__ void ln_prep(const float* __restrict__ in, const float* __restrict__ g,
                        const float* __restrict__ be, float* __restrict__ outf,
                        __bf16* __restrict__ oh, __bf16* __restrict__ ol) {
    const int row = blockIdx.x;
    const int tid = threadIdx.x;
    const float* x = in + (size_t)row * DDIM;
    float s = 0.f, s2 = 0.f;
    for (int i = tid; i < DDIM; i += 256) { float v = x[i]; s += v; s2 += v * v; }
    for (int off = 16; off > 0; off >>= 1) {
        s  += __shfl_xor(s,  off);
        s2 += __shfl_xor(s2, off);
    }
    __shared__ float red[16];
    const int w = tid >> 5, ln = tid & 31;
    if (ln == 0) { red[w * 2] = s; red[w * 2 + 1] = s2; }
    __syncthreads();
    if (tid == 0) {
        float S = 0.f, S2 = 0.f;
        for (int i = 0; i < 8; ++i) { S += red[2 * i]; S2 += red[2 * i + 1]; }
        red[0] = S; red[1] = S2;
    }
    __syncthreads();
    const float mu  = red[0] * (1.0f / DDIM);
    const float var = red[1] * (1.0f / DDIM) - mu * mu;
    const float rs  = rsqrtf(var + 1e-5f);
    for (int i = tid; i < DDIM; i += 256) {
        float y = (x[i] - mu) * rs * g[i] + be[i];
        if (outf) outf[(size_t)row * DDIM + i] = y;
        if (oh) {
            __bf16 h, l;
            split_bf16(y, h, l);
            oh[(size_t)row * DDIM + i] = h;
            ol[(size_t)row * DDIM + i] = l;
        }
    }
}

// ---------------- prep: split W fp32 -> bf16 hi/lo ----------------
__global__ void w_split(const float* __restrict__ W, __bf16* __restrict__ wh,
                        __bf16* __restrict__ wl) {
    const size_t i = ((size_t)blockIdx.x * 256 + threadIdx.x) * 4;
    float4 v = *(const float4*)(W + i);
    __bf16 h, l;
    split_bf16(v.x, h, l); wh[i + 0] = h; wl[i + 0] = l;
    split_bf16(v.y, h, l); wh[i + 1] = h; wl[i + 1] = l;
    split_bf16(v.z, h, l); wh[i + 2] = h; wl[i + 2] = l;
    split_bf16(v.w, h, l); wh[i + 3] = h; wl[i + 3] = l;
}

// ---------------- fused: GEMM1 + residual -> LDS -> GEMM2 -> softmax ----------------
__global__ __launch_bounds__(256) void fused_kernel(
    const float* __restrict__ O, const float* __restrict__ pemb,
    const float* __restrict__ bias,
    const __bf16* __restrict__ Wh, const __bf16* __restrict__ Wl,
    const __bf16* __restrict__ Eh, const __bf16* __restrict__ El,
    const float* __restrict__ lnep, float* __restrict__ out) {

    __shared__ __align__(32) char smem[2048 + 16 * DDIM * 2 * 2]; // 51200 B
    __bf16* sAh = (__bf16*)smem;             // [16][32] stage, hi
    __bf16* sAl = sAh + 512;                 // [16][32] stage, lo
    __bf16* sDh = (__bf16*)(smem + 2048);    // [16][768] dense hi
    __bf16* sDl = sDh + 16 * DDIM;           // [16][768] dense lo
    float*  sSc = (float*)(smem + 2048);     // [16][512] scores (reused)

    const int m0   = blockIdx.x * 16;
    const int p0   = m0 & (PDIM - 1);
    const int tid  = threadIdx.x;
    const int w    = tid >> 5;
    const int lane = tid & 31;
    const int ln16 = lane & 15;
    const int rowOff = (lane >> 4) << 3;        // C/D layout: row = r + 0/8
    const int kselA  = (lane >= 16) ? 16 : 0;   // A chunk select
    const int kselB  = (lane >= 16) ? 16 : 0;   // B K-half select

    // ---- Phase 1: dense[16,768] = [lnep | O] @ W^T  (K = 1536) ----
    v8f acc[6];
#pragma unroll
    for (int t = 0; t < 6; ++t) acc[t] = (v8f){0.f,0.f,0.f,0.f,0.f,0.f,0.f,0.f};

    for (int kt = 0; kt < 48; ++kt) {
        { // cooperative stage: 16x32 fp32 tile -> bf16 hi/lo in LDS
            const int e = tid * 2;
            const int r = e >> 5, c = e & 31;
            const float* src = (kt < 24)
                ? (lnep + (size_t)(p0 + r) * DDIM + kt * 32 + c)
                : (O + (size_t)(m0 + r) * DDIM + (kt - 24) * 32 + c);
            float2 v = *(const float2*)src;
            __bf16 h0, l0, h1, l1;
            split_bf16(v.x, h0, l0);
            split_bf16(v.y, h1, l1);
            sAh[r * 32 + c] = h0; sAh[r * 32 + c + 1] = h1;
            sAl[r * 32 + c] = l0; sAl[r * 32 + c + 1] = l1;
        }
        __syncthreads();
        const v16bf ah = load_a_frag(sAh + ln16 * 32, kselA);
        const v16bf al = load_a_frag(sAl + ln16 * 32, kselA);
        const int kg = kt * 32 + kselB;
#pragma unroll
        for (int t = 0; t < 6; ++t) {
            const int n = w * 96 + t * 16 + ln16;
            const v16bf bh = *(const v16bf*)(Wh + (size_t)n * D2 + kg);
            const v16bf bl = *(const v16bf*)(Wl + (size_t)n * D2 + kg);
            acc[t] = WMMA_BF16(ah, bh, acc[t]);
            acc[t] = WMMA_BF16(ah, bl, acc[t]);
            acc[t] = WMMA_BF16(al, bh, acc[t]);
        }
        __syncthreads();
    }

    // epilogue: + bias + pemb + O residual, split to bf16 hi/lo in LDS
#pragma unroll
    for (int t = 0; t < 6; ++t) {
        const int col = w * 96 + t * 16 + ln16;
        const float bc = bias[col];
#pragma unroll
        for (int r = 0; r < 8; ++r) {
            const int row = r + rowOff;
            float v = acc[t][r] + bc
                    + pemb[(size_t)(p0 + row) * DDIM + col]
                    + O[(size_t)(m0 + row) * DDIM + col];
            __bf16 h, l;
            split_bf16(v, h, l);
            sDh[row * DDIM + col] = h;
            sDl[row * DDIM + col] = l;
        }
    }
    __syncthreads();

    // ---- Phase 2: scores[16,512] = dense @ ecol^T  (K = 768) ----
    v8f acc2[4];
#pragma unroll
    for (int t = 0; t < 4; ++t) acc2[t] = (v8f){0.f,0.f,0.f,0.f,0.f,0.f,0.f,0.f};

    for (int kt = 0; kt < 24; ++kt) {
        const v16bf ah = load_a_frag(sDh + ln16 * DDIM + kt * 32, kselA);
        const v16bf al = load_a_frag(sDl + ln16 * DDIM + kt * 32, kselA);
        const int kg = kt * 32 + kselB;
#pragma unroll
        for (int t = 0; t < 4; ++t) {
            const int c = w * 64 + t * 16 + ln16;
            const v16bf bh = *(const v16bf*)(Eh + (size_t)c * DDIM + kg);
            const v16bf bl = *(const v16bf*)(El + (size_t)c * DDIM + kg);
            acc2[t] = WMMA_BF16(ah, bh, acc2[t]);
            acc2[t] = WMMA_BF16(ah, bl, acc2[t]);
            acc2[t] = WMMA_BF16(al, bh, acc2[t]);
        }
    }
    __syncthreads();   // dense LDS now dead; reuse region for scores

#pragma unroll
    for (int t = 0; t < 4; ++t) {
        const int col = w * 64 + t * 16 + ln16;
#pragma unroll
        for (int r = 0; r < 8; ++r) {
            sSc[(r + rowOff) * CDIM + col] = acc2[t][r];
        }
    }
    __syncthreads();

    // ---- softmax over C=512: wave w handles rows 2w, 2w+1 ----
    for (int rr = 0; rr < 2; ++rr) {
        const int row = w * 2 + rr;
        float vals[16];
        float mx = -3.0e38f;
#pragma unroll
        for (int i = 0; i < 16; ++i) {
            vals[i] = sSc[row * CDIM + lane + i * 32];
            mx = fmaxf(mx, vals[i]);
        }
        for (int off = 16; off > 0; off >>= 1) mx = fmaxf(mx, __shfl_xor(mx, off));
        float sum = 0.f;
#pragma unroll
        for (int i = 0; i < 16; ++i) { vals[i] = expf(vals[i] - mx); sum += vals[i]; }
        for (int off = 16; off > 0; off >>= 1) sum += __shfl_xor(sum, off);
        const float inv = 1.0f / sum;
        float* orow = out + (size_t)(m0 + row) * CDIM;
#pragma unroll
        for (int i = 0; i < 16; ++i) orow[lane + i * 32] = vals[i] * inv;
    }
}

extern "C" void kernel_launch(void* const* d_in, const int* in_sizes, int n_in,
                              void* d_out, int out_size, void* d_ws, size_t ws_size,
                              hipStream_t stream) {
    const float* O      = (const float*)d_in[0];
    const float* colemb = (const float*)d_in[1];
    const float* pemb   = (const float*)d_in[2];
    const float* W      = (const float*)d_in[3];
    const float* b      = (const float*)d_in[4];
    const float* g1     = (const float*)d_in[5];
    const float* b1     = (const float*)d_in[6];
    const float* g2     = (const float*)d_in[7];
    const float* b2     = (const float*)d_in[8];
    float* out = (float*)d_out;

    char* ws = (char*)d_ws;
    __bf16* Wh   = (__bf16*)(ws + 0);          // [768][1536] bf16: 2,359,296 B
    __bf16* Wl   = (__bf16*)(ws + 2359296);    // same
    __bf16* Eh   = (__bf16*)(ws + 4718592);    // [512][768] bf16: 786,432 B
    __bf16* El   = (__bf16*)(ws + 5505024);    // same
    float*  lnep = (float*)(ws + 6291456);     // [128][768] f32: 393,216 B

    ln_prep<<<PDIM, 256, 0, stream>>>(pemb,   g1, b1, lnep, nullptr, nullptr);
    ln_prep<<<CDIM, 256, 0, stream>>>(colemb, g2, b2, nullptr, Eh, El);
    w_split<<<(DDIM * D2) / (256 * 4), 256, 0, stream>>>(W, Wh, Wl);
    fused_kernel<<<(256 * PDIM) / 16, 256, 0, stream>>>(O, pemb, b, Wh, Wl, Eh, El, lnep, out);
}
